// Sampler_10453950398946
// MI455X (gfx1250) — matrix operand order repaired and checked by reading, not compiled, not run on hardware
//
#include <hip/hip_runtime.h>
#include <hip/hip_bf16.h>
#include <math.h>

// Problem constants (match reference)
constexpr int NN  = 256;      // sampled tokens
constexpr int DD  = 1024;     // d_model
constexpr int VV  = 128000;   // vocab

typedef __attribute__((ext_vector_type(8)))  float  v8f;
typedef __attribute__((ext_vector_type(16))) __bf16 v16bf;
typedef __attribute__((ext_vector_type(8)))  __bf16 v8bf;
typedef __attribute__((ext_vector_type(4)))  __bf16 v4bf;

// ---------------------------------------------------------------------------
// Kernel 0: gather last-token rows AND pre-split them into bf16 hi/lo planes:
//   x = hi + lo,  hi = bf16(x), lo = bf16(x - hi)   (residual ~2^-18 relative)
// hg_hi / hg_lo: [N, D] bf16 planes in workspace (0.5 MB each).
// ---------------------------------------------------------------------------
__global__ void gather_split(const float* __restrict__ hidden,
                             const int* __restrict__ idx,
                             __bf16* __restrict__ hg_hi,
                             __bf16* __restrict__ hg_lo) {
    const int r = blockIdx.x;          // 0..255
    const int t = threadIdx.x;         // 0..255, one float4 each (1024 f32/row)
    const int src = idx[r];
    float4 v = ((const float4*)(hidden + (size_t)src * DD))[t];
    v4bf hi, lo;
    hi.x = (__bf16)v.x; lo.x = (__bf16)(v.x - (float)hi.x);
    hi.y = (__bf16)v.y; lo.y = (__bf16)(v.y - (float)hi.y);
    hi.z = (__bf16)v.z; lo.z = (__bf16)(v.z - (float)hi.z);
    hi.w = (__bf16)v.w; lo.w = (__bf16)(v.w - (float)hi.w);
    ((v4bf*)(hg_hi + (size_t)r * DD))[t] = hi;
    ((v4bf*)(hg_lo + (size_t)r * DD))[t] = lo;
}

// ---------------------------------------------------------------------------
// Kernel 1: logits[n,v] = sum_d hg[n,d] * emb[v,d]
// BF16-split WMMA: x*y ~= xh*yh + xh*yl + xl*yh  via V_WMMA_F32_16X16X32_BF16.
// Block tile: M=256 x V=128 (embedding streams from HBM exactly once).
// 512 threads = 16 waves = 2 (M-halves of 128 rows) x 8 (V-groups of 16 cols);
// each wave owns 8 M-subtiles -> 64 accumulator VGPRs, 4 waves/SIMD for
// latency hiding.  B chunk register-staged one iteration ahead (hides HBM
// latency); A fragments double-buffered tile-to-tile (hides LDS latency).
// ---------------------------------------------------------------------------
#define KC  32
#define LDA 40   // padded pitch in bf16: 80B = 20 banks (conflict-free), 16B-aligned

__device__ __forceinline__ v16bf cat8(v8bf a, v8bf b) {
    return __builtin_shufflevector(a, b, 0,1,2,3,4,5,6,7,8,9,10,11,12,13,14,15);
}
__device__ __forceinline__ v16bf ldfrag(const __bf16* base) {
    // two ds_load_b128 into consecutive VGPRs (K 0..7 | 8..15 relative to base)
    return cat8(*(const v8bf*)(base), *(const v8bf*)(base + 8));
}

__global__ __launch_bounds__(512, 1)
void gemm_logits(const __bf16* __restrict__ hg_hi,
                 const __bf16* __restrict__ hg_lo,
                 const float* __restrict__ emb,
                 float* __restrict__ logits) {
    __shared__ __bf16 sAhi[NN  * LDA];   // 20 KB
    __shared__ __bf16 sAlo[NN  * LDA];   // 20 KB
    __shared__ __bf16 sBhi[128 * LDA];   // 10 KB
    __shared__ __bf16 sBlo[128 * LDA];   // 10 KB

    const int tid  = threadIdx.x;
    const int lane = tid & 31;
    const int wave = tid >> 5;              // 0..15
    const int mg   = (wave & 1) * 128;      // M half: rows [mg, mg+128)
    const int vg   = wave >> 1;             // V group: cols [vg*16, vg*16+16)
    const int v0   = blockIdx.x * 128;      // vocab tile base
    const int nlo  = lane & 15;             // 0..15
    const bool lohalf = (lane < 16);
    const int koffa = lohalf ? 0 : 16;      // A: K0..15 vs K16..31
    const int koffb = lohalf ? 0 : 16;      // B: K0..15 vs K16..31

    v8f acc[8];
    #pragma unroll
    for (int t = 0; t < 8; ++t) acc[t] = (v8f)0.0f;

    // ---- prologue: stage first B chunk (embedding f32) in registers
    float4 stB[2];
    #pragma unroll
    for (int j = 0; j < 2; ++j) {
        int f4 = tid + j * 512;
        int vr = f4 >> 3;
        int c4 = (f4 & 7) << 2;
        stB[j] = *(const float4*)(emb + (size_t)(v0 + vr) * DD + c4);
    }

    for (int ko = 0; ko < DD; ko += KC) {
        // ---- A: straight bf16 copy, global(L2) -> LDS. 2 b128 per plane/thread
        #pragma unroll
        for (int j = 0; j < 2; ++j) {
            int f4 = tid + j * 512;          // 1024 b128 per plane
            int row = f4 >> 2;               // 4 b128 per 32-bf16 row
            int c8  = (f4 & 3) << 3;
            uint4 vh = *(const uint4*)(hg_hi + (size_t)row * DD + ko + c8);
            uint4 vl = *(const uint4*)(hg_lo + (size_t)row * DD + ko + c8);
            *(uint4*)(sAhi + row * LDA + c8) = vh;
            *(uint4*)(sAlo + row * LDA + c8) = vl;
        }
        // ---- B: split staged f32 chunk -> bf16 hi/lo planes in LDS
        #pragma unroll
        for (int j = 0; j < 2; ++j) {
            int f4 = tid + j * 512;
            int vr = f4 >> 3;
            int c4 = (f4 & 7) << 2;
            float4 v = stB[j];
            v4bf hi, lo;
            hi.x = (__bf16)v.x; lo.x = (__bf16)(v.x - (float)hi.x);
            hi.y = (__bf16)v.y; lo.y = (__bf16)(v.y - (float)hi.y);
            hi.z = (__bf16)v.z; lo.z = (__bf16)(v.z - (float)hi.z);
            hi.w = (__bf16)v.w; lo.w = (__bf16)(v.w - (float)hi.w);
            *(v4bf*)(sBhi + vr * LDA + c4) = hi;
            *(v4bf*)(sBlo + vr * LDA + c4) = lo;
        }
        __syncthreads();

        // ---- stage next B chunk from HBM; completes during the WMMA phase
        if (ko + KC < DD) {
            #pragma unroll
            for (int j = 0; j < 2; ++j) {
                int f4 = tid + j * 512;
                int vr = f4 >> 3;
                int c4 = (f4 & 7) << 2;
                stB[j] = *(const float4*)(emb + (size_t)(v0 + vr) * DD + ko + KC + c4);
            }
            __builtin_prefetch(emb + (size_t)(v0 + (tid >> 2)) * DD + ko + 2 * KC, 0, 1);
        }

        // ---- one 32-K WMMA step per chunk
        // B fragments (32x16): lane = column n, 16 contiguous K per half-wave
        const __bf16* bh = sBhi + (vg * 16 + nlo) * LDA + koffb;
        const __bf16* bl = sBlo + (vg * 16 + nlo) * LDA + koffb;
        v16bf bhi = cat8(*(const v8bf*)(bh), *(const v8bf*)(bh + 8));
        v16bf blo = cat8(*(const v8bf*)(bl), *(const v8bf*)(bl + 8));

        // A fragments (16x32), double-buffered across tiles
        v16bf ahi = ldfrag(sAhi + (mg + nlo) * LDA + koffa);
        v16bf alo = ldfrag(sAlo + (mg + nlo) * LDA + koffa);
        #pragma unroll
        for (int t = 0; t < 8; ++t) {
            v16bf ahiN = ahi, aloN = alo;
            if (t < 7) {
                ahiN = ldfrag(sAhi + (mg + (t + 1) * 16 + nlo) * LDA + koffa);
                aloN = ldfrag(sAlo + (mg + (t + 1) * 16 + nlo) * LDA + koffa);
            }
            acc[t] = __builtin_amdgcn_wmma_f32_16x16x32_bf16(
                false, ahi, false, bhi, (short)0, acc[t], false, false);
            acc[t] = __builtin_amdgcn_wmma_f32_16x16x32_bf16(
                false, ahi, false, blo, (short)0, acc[t], false, false);
            acc[t] = __builtin_amdgcn_wmma_f32_16x16x32_bf16(
                false, alo, false, bhi, (short)0, acc[t], false, false);
            ahi = ahiN; alo = aloN;
        }
        __syncthreads();
    }

    // ---- epilogue: C 16x16 layout -> VGPR r holds M = r (+8 for upper lanes)
    const int col = v0 + vg * 16 + nlo;
    const int hi8 = lohalf ? 0 : 8;
    #pragma unroll
    for (int t = 0; t < 8; ++t) {
        #pragma unroll
        for (int r = 0; r < 8; ++r) {
            int row = mg + t * 16 + r + hi8;
            logits[(size_t)row * VV + col] = acc[t][r];
        }
    }
}

// ---------------------------------------------------------------------------
// Kernel 2: per-row (one block per row, 256 threads, L2-resident data):
//   softmax(logits/T)  ->  top-p via binary-searched threshold  ->  renorm
//   ->  Gumbel-max sample.  All in place in d_out.
// ---------------------------------------------------------------------------
__device__ __forceinline__ float block_max(float v, float* red) {
    const int lane = threadIdx.x & 31, wave = threadIdx.x >> 5;
    #pragma unroll
    for (int o = 16; o; o >>= 1) v = fmaxf(v, __shfl_xor(v, o, 32));
    __syncthreads();
    if (lane == 0) red[wave] = v;
    __syncthreads();
    float r = red[0];
    #pragma unroll
    for (int w = 1; w < 8; ++w) r = fmaxf(r, red[w]);
    return r;
}

__device__ __forceinline__ float block_sum(float v, float* red) {
    const int lane = threadIdx.x & 31, wave = threadIdx.x >> 5;
    #pragma unroll
    for (int o = 16; o; o >>= 1) v += __shfl_xor(v, o, 32);
    __syncthreads();
    if (lane == 0) red[wave] = v;
    __syncthreads();
    float r = red[0];
    #pragma unroll
    for (int w = 1; w < 8; ++w) r += red[w];
    return r;
}

__device__ __forceinline__ float hash_u01(unsigned x) {
    x ^= x >> 17; x *= 0xed5ad4bbu;
    x ^= x >> 11; x *= 0xac4c1b51u;
    x ^= x >> 15; x *= 0x31848babu;
    x ^= x >> 14;
    return ((float)(x >> 8) + 0.5f) * (1.0f / 16777216.0f);
}

__global__ __launch_bounds__(256)
void softmax_topp_sample(float* __restrict__ out,           // [N*V] probs + [N] tokens
                         const float* __restrict__ temps,
                         const float* __restrict__ topps) {
    const int row = blockIdx.x;
    const int tid = threadIdx.x;
    float* p = out + (size_t)row * VV;

    __shared__ float red[8];
    __shared__ float s_lo, s_hi, s_S;
    __shared__ float rv[8];
    __shared__ int   ri[8];

    const float invT = 1.0f / temps[row];
    const float topp = topps[row];

    // pass 1: row max of scaled logits
    float m = -3.4e38f;
    for (int i = tid; i < VV; i += 256) m = fmaxf(m, p[i] * invT);
    m = block_max(m, red);

    // pass 2: sum exp
    float Z = 0.0f;
    for (int i = tid; i < VV; i += 256) Z += __expf(p[i] * invT - m);
    Z = block_sum(Z, red);
    const float rZ = 1.0f / Z;

    // pass 3: write probabilities (in place over logits)
    for (int i = tid; i < VV; i += 256) p[i] = __expf(p[i] * invT - m) * rZ;
    __syncthreads();

    // binary search for the top-p threshold t*:
    // keep p_i iff sum_{q > p_i} q <= topp; f(x)=sum_{q>x} q is monotone, so
    // lo converges just below the smallest kept prob; f(lo) = kept mass.
    if (tid == 0) { s_lo = 0.0f; s_hi = 1.0f; s_S = 1.0f; }
    __syncthreads();
    for (int it = 0; it < 30; ++it) {
        const float mid = 0.5f * (s_lo + s_hi);
        float S = 0.0f;
        for (int i = tid; i < VV; i += 256) {
            float v = p[i];
            if (v > mid) S += v;
        }
        S = block_sum(S, red);
        if (tid == 0) {
            if (S > topp) { s_lo = mid; s_S = S; }
            else          { s_hi = mid; }
        }
        __syncthreads();
    }
    const float thr = s_lo;
    const float rS  = 1.0f / s_S;

    // mask tail + renormalize (in place)
    for (int i = tid; i < VV; i += 256) {
        float v = p[i];
        p[i] = (v > thr) ? v * rS : 0.0f;
    }
    __syncthreads();

    // Gumbel-max sample over renormalized probs
    float best = -3.4e38f;
    int bi = 0;
    for (int i = tid; i < VV; i += 256) {
        float v = p[i];
        unsigned seed = (unsigned)row * 0x9E3779B1u ^ (unsigned)i * 0x85EBCA77u ^ 42u;
        float u = hash_u01(seed);
        float g = -__logf(-__logf(u));
        float sc = __logf(v + 1e-20f) + g;
        if (sc > best) { best = sc; bi = i; }
    }
    const int lane = tid & 31, wave = tid >> 5;
    #pragma unroll
    for (int o = 16; o; o >>= 1) {
        float ov = __shfl_xor(best, o, 32);
        int   oi = __shfl_xor(bi,   o, 32);
        if (ov > best || (ov == best && oi < bi)) { best = ov; bi = oi; }
    }
    if (lane == 0) { rv[wave] = best; ri[wave] = bi; }
    __syncthreads();
    if (tid == 0) {
        float b = rv[0]; int j = ri[0];
        #pragma unroll
        for (int w = 1; w < 8; ++w)
            if (rv[w] > b || (rv[w] == b && ri[w] < j)) { b = rv[w]; j = ri[w]; }
        out[(size_t)NN * VV + row] = (float)j;
    }
}

// ---------------------------------------------------------------------------
extern "C" void kernel_launch(void* const* d_in, const int* in_sizes, int n_in,
                              void* d_out, int out_size, void* d_ws, size_t ws_size,
                              hipStream_t stream) {
    const float* hidden = (const float*)d_in[0];   // [T, D]
    const float* emb    = (const float*)d_in[1];   // [V, D]
    const float* temps  = (const float*)d_in[2];   // [N]
    const float* topps  = (const float*)d_in[3];   // [N]
    const int*   idx    = (const int*)  d_in[4];   // [N]
    float* out = (float*)d_out;                    // [N*V] probs ++ [N] tokens

    __bf16* hg_hi = (__bf16*)d_ws;                 // [N, D] bf16 hi plane
    __bf16* hg_lo = hg_hi + (size_t)NN * DD;       // [N, D] bf16 lo plane

    gather_split<<<NN, 256, 0, stream>>>(hidden, idx, hg_hi, hg_lo);
    gemm_logits<<<VV / 128, 512, 0, stream>>>(hg_hi, hg_lo, emb, out);
    softmax_topp_sample<<<NN, 256, 0, stream>>>(out, temps, topps);
}
